// SpatialGraphConv_45440753992501
// MI455X (gfx1250) — matrix operand order repaired and checked by reading, not compiled, not run on hardware
//
#include <hip/hip_runtime.h>
#include <hip/hip_bf16.h>

// ---------------------------------------------------------------------------
// Chebyshev spatial graph conv on MI455X (gfx1250), bf16 WMMA path.
// B=32, T=48 (BT=1536), N=512, C_IN=32, C_OUT=64, K=3
// X buffers and theta are kept in WMMA *B-fragment-swizzled* bf16 layout so
// every fragment load is a pair of b128 LDS reads (no per-lane u16 gathers).
// ---------------------------------------------------------------------------

typedef __attribute__((ext_vector_type(16))) __bf16 v16bf;
typedef __attribute__((ext_vector_type(8)))  float  v8f;
typedef __attribute__((ext_vector_type(8)))  int    v8i;
typedef int v4i_gcc __attribute__((vector_size(16)));   // matches async-LDS builtin param

#define BT    1536
#define NN    512
#define CIN   32
#define COUT  64
#define ROWS_PER_BLOCK 128
#define ROW_TILES (NN / ROWS_PER_BLOCK)   // 4
#define XSW_DW_PER_BT 8192                // 16 ktiles * 2 ctiles * 32 lanes * 8 dwords

#if defined(__HIP_DEVICE_COMPILE__) && __has_builtin(__builtin_amdgcn_global_load_async_to_lds_b128)
#define HAVE_ASYNC 1
#else
#define HAVE_ASYNC 0
#endif

__device__ __forceinline__ unsigned short f2bf(float f) {
    unsigned u = __builtin_bit_cast(unsigned, f);
    unsigned r = (u + 0x7FFFu + ((u >> 16) & 1u)) >> 16;   // round-to-nearest-even
    return (unsigned short)r;
}

// 16-byte global -> LDS copy (async on CDNA5, falls back to reg staging).
__device__ __forceinline__ void copy16(void* l, const void* g) {
#if HAVE_ASYNC
    __builtin_amdgcn_global_load_async_to_lds_b128(
        (__attribute__((address_space(1))) v4i_gcc*)g,
        (__attribute__((address_space(3))) v4i_gcc*)l, 0, 0);
#else
    *(uint4*)l = *(const uint4*)g;
#endif
}

// A fragment (16x32 bf16, M x K) from a row-major LDS tile, row stride ldh halves.
// lanes 0-15 -> M=lane, K{0..15} in v0-7; lanes 16-31 -> K{16..31} (contiguous, matches B swizzle).
__device__ __forceinline__ v16bf load_A_frag(const unsigned short* base, int ldh) {
    int lane = threadIdx.x & 31;
    int half = lane >> 4;
    int m    = lane & 15;
    const unsigned short* rp = base + m * ldh + half * 16;
    uint4 c0 = *(const uint4*)(rp);        // K = half*16 + 0..7
    uint4 c1 = *(const uint4*)(rp + 8);    // K = half*16 + 8..15
    v8i v;
    v[0] = (int)c0.x; v[1] = (int)c0.y; v[2] = (int)c0.z; v[3] = (int)c0.w;
    v[4] = (int)c1.x; v[5] = (int)c1.y; v[6] = (int)c1.z; v[7] = (int)c1.w;
    return __builtin_bit_cast(v16bf, v);
}

// B fragment (32x16 bf16) from a pre-swizzled LDS slab: lane reads 8 consecutive dwords.
__device__ __forceinline__ v16bf load_B_sw(const unsigned int* slab) {
    int lane = threadIdx.x & 31;
    const uint4* p = (const uint4*)(slab + lane * 8);
    uint4 d0 = p[0];
    uint4 d1 = p[1];
    v8i v;
    v[0] = (int)d0.x; v[1] = (int)d0.y; v[2] = (int)d0.z; v[3] = (int)d0.w;
    v[4] = (int)d1.x; v[5] = (int)d1.y; v[6] = (int)d1.z; v[7] = (int)d1.w;
    return __builtin_bit_cast(v16bf, v);
}

// ---------------------------------------------------------------------------
// fp32 -> bf16 conversion: x -> B-swizzled X0, cheb -> row-major, theta -> B-swizzled
// ---------------------------------------------------------------------------
__global__ __launch_bounds__(256) void cheb_convert(
    const float* __restrict__ x, const float* __restrict__ cheb,
    const float* __restrict__ theta,
    unsigned int*   __restrict__ xsw,     // BT * 8192 dwords
    unsigned short* __restrict__ chebbf,  // 3*512*512 halves
    unsigned int*   __restrict__ thsw,    // 3 * 1024 dwords
    int nxw, int ncv, int ntw)
{
    int i      = blockIdx.x * blockDim.x + threadIdx.x;
    int stride = gridDim.x * blockDim.x;

    for (int idx = i; idx < nxw; idx += stride) {           // 12,582,912 dwords
        int j  = idx & 7;
        int L  = (idx >> 3) & 31;
        int ct = (idx >> 8) & 1;
        int kt = (idx >> 9) & 15;
        int bt = idx >> 13;
        int row = kt * 32 + ((L >> 4) << 4) + j * 2;
        int col = ct * 16 + (L & 15);
        const float* xp = x + ((size_t)bt * NN + row) * CIN + col;
        unsigned lo = f2bf(__builtin_nontemporal_load(xp));
        unsigned hi = f2bf(__builtin_nontemporal_load(xp + CIN));
        xsw[idx] = lo | (hi << 16);
    }
    for (int idx = i; idx < ncv; idx += stride)             // 786,432 halves
        chebbf[idx] = f2bf(__builtin_nontemporal_load(cheb + idx));
    for (int idx = i; idx < ntw; idx += stride) {           // 3,072 dwords
        int j  = idx & 7;
        int L  = (idx >> 3) & 31;
        int ct = (idx >> 8) & 3;
        int k  = idx >> 10;
        int row = ((L >> 4) << 4) + j * 2;                  // c_in
        int col = ct * 16 + (L & 15);                       // c_out
        const float* tp = theta + ((size_t)k * CIN + row) * COUT + col;
        thsw[idx] = (unsigned)f2bf(tp[0]) | ((unsigned)f2bf(tp[COUT]) << 16);
    }
}

// ---------------------------------------------------------------------------
// One Chebyshev order: Xk = chebK @ Xin ; out (+)= Xk @ thetaK ; ReLU on last.
// grid = (ROW_TILES, BT), block = 256 (8 waves); wave w owns 16 output rows.
// ---------------------------------------------------------------------------
__global__ __launch_bounds__(256) void cheb_pass(
    const unsigned short* __restrict__ chebK,   // 512*512 bf16 row-major
    const unsigned int*   __restrict__ thetaKW, // 1024 dwords, B-swizzled
    const unsigned int*   __restrict__ XinW,    // BT*8192 dwords, B-swizzled
    unsigned int*         __restrict__ XoutW,   // BT*8192 dwords, B-swizzled
    float*                __restrict__ out,     // BT*512*64 f32
    int first, int last)
{
    __shared__ __align__(16) unsigned int   BstW[XSW_DW_PER_BT];          // 32 KB X_{k-1}
    __shared__ __align__(16) unsigned short Ast[2][ROWS_PER_BLOCK * 32];  // 16 KB cheb pingpong
    __shared__ __align__(16) unsigned int   ThW[1024];                    //  4 KB theta_k
    __shared__ __align__(16) unsigned short Xst[ROWS_PER_BLOCK * CIN];    //  8 KB X_k restage

    const int bt      = blockIdx.y;
    const int rowbase = blockIdx.x * ROWS_PER_BLOCK;
    const int tid     = threadIdx.x;
    const int wave    = tid >> 5;
    const int lane    = tid & 31;
    const int half    = lane >> 4;
    const int ncl     = lane & 15;

    // theta_k -> LDS (256 uint4, 1 per thread)
    copy16(ThW + tid * 4, thetaKW + tid * 4);
    // X_{k-1}(bt) -> LDS (2048 uint4, 8 per thread) : already B-swizzled
    {
        const unsigned int* src = XinW + (size_t)bt * XSW_DW_PER_BT;
#pragma unroll
        for (int i = 0; i < 8; ++i)
            copy16(BstW + (tid + i * 256) * 4, src + (tid + i * 256) * 4);
    }
    // cheb chunk 0 -> Ast[0]  (rows [rowbase,+128) x cols [0,32) : 2 uint4 / thread)
    {
        const unsigned short* s0 = chebK + (size_t)(rowbase +      (tid >> 2)) * NN + (tid & 3) * 8;
        const unsigned short* s1 = chebK + (size_t)(rowbase + 64 + (tid >> 2)) * NN + (tid & 3) * 8;
        copy16(Ast[0] + tid * 8, s0);
        copy16(Ast[0] + (tid + 256) * 8, s1);
    }

    // -------- propagation: acc(16x32) = cheb[rows] @ X_{k-1} --------
    v8f c0 = {}, c1 = {};
    for (int kt = 0; kt < 16; ++kt) {
        if (kt + 1 < 16) {   // stage next cheb chunk into the other buffer
            int kk = (kt + 1) * 32;
            const unsigned short* s0 = chebK + (size_t)(rowbase +      (tid >> 2)) * NN + kk + (tid & 3) * 8;
            const unsigned short* s1 = chebK + (size_t)(rowbase + 64 + (tid >> 2)) * NN + kk + (tid & 3) * 8;
            copy16(Ast[(kt + 1) & 1] + tid * 8, s0);
            copy16(Ast[(kt + 1) & 1] + (tid + 256) * 8, s1);
#if HAVE_ASYNC
            asm volatile("s_wait_asynccnt 0x2" ::: "memory");  // newest chunk may stay in flight
#endif
        } else {
#if HAVE_ASYNC
            asm volatile("s_wait_asynccnt 0x0" ::: "memory");
#endif
        }
        __syncthreads();

        v16bf a  = load_A_frag(Ast[kt & 1] + wave * 16 * 32, 32);
        v16bf b0 = load_B_sw(BstW + (kt * 2 + 0) * 256);
        v16bf b1 = load_B_sw(BstW + (kt * 2 + 1) * 256);
        c0 = __builtin_amdgcn_wmma_f32_16x16x32_bf16(false, a, false, b0,
                                                     (short)0, c0, false, false);
        c1 = __builtin_amdgcn_wmma_f32_16x16x32_bf16(false, a, false, b1,
                                                     (short)0, c1, false, false);
        __syncthreads();
    }

    // -------- write X_k: B-swizzled to global, row-major restage in LDS --------
    if (!last) {   // X_3 is never consumed
        unsigned int* xg = XoutW + (size_t)bt * XSW_DW_PER_BT;
        int kt2 = (rowbase + wave * 16) >> 5;          // ktile this wave contributes to
        int Lt  = ((wave & 1) << 4) + ncl;             // target lane slot in the slab
        uint4 q0, q1;
        q0.x = (unsigned)f2bf(c0[0]) | ((unsigned)f2bf(c0[1]) << 16);
        q0.y = (unsigned)f2bf(c0[2]) | ((unsigned)f2bf(c0[3]) << 16);
        q0.z = (unsigned)f2bf(c0[4]) | ((unsigned)f2bf(c0[5]) << 16);
        q0.w = (unsigned)f2bf(c0[6]) | ((unsigned)f2bf(c0[7]) << 16);
        q1.x = (unsigned)f2bf(c1[0]) | ((unsigned)f2bf(c1[1]) << 16);
        q1.y = (unsigned)f2bf(c1[2]) | ((unsigned)f2bf(c1[3]) << 16);
        q1.z = (unsigned)f2bf(c1[4]) | ((unsigned)f2bf(c1[5]) << 16);
        q1.w = (unsigned)f2bf(c1[6]) | ((unsigned)f2bf(c1[7]) << 16);
        *(uint4*)(xg + ((kt2 * 2 + 0) * 32 + Lt) * 8 + 4 * half) = q0;
        *(uint4*)(xg + ((kt2 * 2 + 1) * 32 + Lt) * 8 + 4 * half) = q1;
    }
#pragma unroll
    for (int r = 0; r < 8; ++r) {                      // row-major restage for A fragment
        int m = r + 8 * half;                          // C/D layout: M = r + 8*(lane>=16)
        Xst[(wave * 16 + m) * CIN + ncl]      = f2bf(c0[r]);
        Xst[(wave * 16 + m) * CIN + 16 + ncl] = f2bf(c1[r]);
    }
    __syncthreads();

    // -------- projection: out (+)= X_k(16x32) @ theta_k(32x64) --------
    v16bf xa = load_A_frag(Xst + wave * 16 * CIN, CIN);
    float* op = out + (size_t)bt * NN * COUT + (size_t)(rowbase + wave * 16) * COUT;
#pragma unroll
    for (int t = 0; t < 4; ++t) {
        v16bf tb = load_B_sw(ThW + t * 256);
        v8f cc = {};
        if (!first) {
#pragma unroll
            for (int r = 0; r < 8; ++r)
                cc[r] = op[(r + 8 * half) * COUT + t * 16 + ncl];
        }
        v8f d = __builtin_amdgcn_wmma_f32_16x16x32_bf16(false, xa, false, tb,
                                                        (short)0, cc, false, false);
#pragma unroll
        for (int r = 0; r < 8; ++r) {
            float v = d[r];
            if (last) {
                v = v > 0.0f ? v : 0.0f;
                __builtin_nontemporal_store(v, &op[(r + 8 * half) * COUT + t * 16 + ncl]);
            } else {
                op[(r + 8 * half) * COUT + t * 16 + ncl] = v;
            }
        }
    }
}

// ---------------------------------------------------------------------------
extern "C" void kernel_launch(void* const* d_in, const int* in_sizes, int n_in,
                              void* d_out, int out_size, void* d_ws, size_t ws_size,
                              hipStream_t stream) {
    const float* x     = (const float*)d_in[0];   // (32,48,512,32)
    const float* cheb  = (const float*)d_in[1];   // (3,512,512)
    const float* theta = (const float*)d_in[2];   // (3,32,64)
    float* out = (float*)d_out;                   // (32,48,512,64)

    const size_t nxw = (size_t)BT * XSW_DW_PER_BT;   // 12,582,912 dwords
    const size_t ncv = (size_t)3 * NN * NN;          //    786,432 halves
    const size_t ntw = (size_t)3 * 1024;             //      3,072 dwords

    char* ws = (char*)d_ws;
    unsigned int*   xsw0   = (unsigned int*)(ws);
    unsigned int*   xsw1   = (unsigned int*)(ws + nxw * 4);
    unsigned short* chebbf = (unsigned short*)(ws + nxw * 8);
    unsigned int*   thsw   = (unsigned int*)(ws + nxw * 8 + ncv * 2);

    cheb_convert<<<4096, 256, 0, stream>>>(x, cheb, theta, xsw0, chebbf, thsw,
                                           (int)nxw, (int)ncv, (int)ntw);

    dim3 grid(ROW_TILES, BT);
    // k = 0: X1 = T0 @ X0 ; out  = X1 @ th0
    cheb_pass<<<grid, 256, 0, stream>>>(chebbf + 0 * NN * NN, thsw + 0 * 1024,
                                        xsw0, xsw1, out, /*first=*/1, /*last=*/0);
    // k = 1: X2 = T1 @ X1 ; out += X2 @ th1
    cheb_pass<<<grid, 256, 0, stream>>>(chebbf + 1 * NN * NN, thsw + 1 * 1024,
                                        xsw1, xsw0, out, /*first=*/0, /*last=*/0);
    // k = 2: X3 = T2 @ X2 ; out = relu(out + X3 @ th2)
    cheb_pass<<<grid, 256, 0, stream>>>(chebbf + 2 * NN * NN, thsw + 2 * 1024,
                                        xsw0, xsw1, out, /*first=*/0, /*last=*/1);
}
